// BottleneckVQa_33466385170758
// MI455X (gfx1250) — compile-verified, matrix operand order checked by reading, not run on hardware
//
#include <hip/hip_runtime.h>
#include <hip/hip_bf16.h>

#if defined(__has_builtin)
#if __has_builtin(__builtin_amdgcn_global_load_async_to_lds_b128) && \
    __has_builtin(__builtin_amdgcn_s_wait_asynccnt)
#define CDNA5_ASYNC 1
#endif
#endif
#ifndef CDNA5_ASYNC
#define CDNA5_ASYNC 0
#endif

typedef __attribute__((ext_vector_type(16))) _Float16 v16h;
typedef __attribute__((ext_vector_type(8)))  _Float16 v8h;
typedef __attribute__((ext_vector_type(8)))  float    v8f;

// exact param type of __builtin_amdgcn_global_load_async_to_lds_b128 (from diagnostics)
typedef int v4i __attribute__((__vector_size__(4 * sizeof(int))));
typedef __attribute__((address_space(1))) v4i gv4i;   // global (__device__)
typedef __attribute__((address_space(3))) v4i sv4i;   // LDS (__shared__)

__device__ __forceinline__ float gelu_f(float x) {
    return 0.5f * x * (1.0f + erff(x * 0.70710678118654752f));
}

// ---------------- weight cast: OIHW fp32 -> [OC][K] f16 (same layout) -----------------
__global__ void wcast_kernel(const float* __restrict__ w, _Float16* __restrict__ out, int n) {
    int i = blockIdx.x * blockDim.x + threadIdx.x;
    if (i < n) out[i] = (_Float16)w[i];
}

// ---------------- encoder conv1: [B,3,256,256] -> [B,64,128,128], 5x5 s2 p2 + GELU ----
__global__ __launch_bounds__(256) void conv1_kernel(
    const float* __restrict__ x, const float* __restrict__ w,
    _Float16* __restrict__ out, int B)
{
    __shared__ float ws_[64 * 75];
    for (int i = threadIdx.x; i < 64 * 75; i += blockDim.x) ws_[i] = w[i];
    __syncthreads();
    long long idx = (long long)blockIdx.x * blockDim.x + threadIdx.x;
    const long long total = (long long)B * 64 * 128 * 128;
    if (idx >= total) return;
    int ox = (int)(idx % 128); long long t = idx / 128;
    int oy = (int)(t % 128); t /= 128;
    int oc = (int)(t % 64);  int img = (int)(t / 64);
    float acc = 0.f;
    const float* xb = x + (long long)img * 3 * 256 * 256;
    #pragma unroll
    for (int ci = 0; ci < 3; ++ci)
        #pragma unroll
        for (int kh = 0; kh < 5; ++kh) {
            int iy = oy * 2 - 2 + kh;
            if (iy < 0 || iy >= 256) continue;
            #pragma unroll
            for (int kw = 0; kw < 5; ++kw) {
                int ix = ox * 2 - 2 + kw;
                if (ix < 0 || ix >= 256) continue;
                acc += xb[(ci * 256 + iy) * 256 + ix] * ws_[(oc * 3 + ci) * 25 + kh * 5 + kw];
            }
        }
    out[idx] = (_Float16)gelu_f(acc);
}

// ---------------- implicit-GEMM conv via v_wmma_f32_16x16x32_f16 ----------------------
// Tile: M=128 (8 waves x 16 rows), N=64 (4 n-subtiles/wave), K staged in chunks of 64
// (2 WMMA K-steps per stage). A: register double-buffered im2col. B: async
// global->LDS (ping-pong, ASYNCcnt) when the toolchain exposes the gfx1250 builtins,
// else register double-buffered. All dims compile-time; Ktot % 64 == 0 throughout.
template<int KH, int KW, int STRIDE, int PAD, bool GELU,
         int Cin, int Hin, int Win, int OC, int OH, int OW, typename TOUT>
__global__ __launch_bounds__(256) void conv_wmma_kernel(
    const _Float16* __restrict__ in, const _Float16* __restrict__ wB, // wB: [OC][Ktot]
    TOUT* __restrict__ out, int B)
{
    constexpr int MT = 128, NT = 64, KC = 64;
    constexpr int KHW  = KH * KW;
    constexpr int Ktot = Cin * KHW;
    constexpr int OHW  = OH * OW;
    static_assert(Ktot % KC == 0, "K tail not handled");

    __shared__ alignas(16) _Float16 As[MT][KC + 8];
    __shared__ alignas(16) _Float16 Bt[2][NT][KC + 8];   // [buf][col][k], ping-pong

    const int tid  = threadIdx.x;
    const int wave = tid >> 5;
    const int lane = tid & 31;
    const int l    = lane & 15;
    const int hi   = lane >> 4;
    const int kc   = tid & 31;            // fixed K column within half-chunk

    const int Mtotal = B * OHW;
    const int mBase  = blockIdx.x * MT;
    const int ocBase = blockIdx.y * NT;

    // ---- per-row im2col decode, hoisted out of the K loop (rows: it*8 + wave) ----
    int rowBase[16];
    int rowYX[16];
    #pragma unroll
    for (int it = 0; it < 16; ++it) {
        int m = mBase + it * 8 + wave;
        if (m < Mtotal) {
            int img = m / OHW;
            int p   = m - img * OHW;
            int oy  = p / OW;
            int ox  = p - oy * OW;
            rowBase[it] = img * (Cin * Hin * Win);
            rowYX[it]   = ((oy * STRIDE - PAD) << 16) | ((ox * STRIDE - PAD) & 0xffff);
        } else {
            rowBase[it] = 0;
            rowYX[it]   = (int)0x80000000;   // iy0 = -32768 -> always OOB -> zeros
        }
    }

    // ---- B-tile fetch coords: 16 contiguous k per thread for one column ----
    const int bcol = tid >> 2;            // 0..63
    const int bk   = (tid & 3) * 16;      // 0,16,32,48
    const _Float16* wCol = wB + (long long)(ocBase + bcol) * Ktot + bk;

    _Float16 aReg[2][16];

    auto fetchA = [&](int kBase) {
        #pragma unroll
        for (int half = 0; half < 2; ++half) {
            int kk   = kBase + half * 32 + kc;
            int ci   = kk / KHW;
            int rem  = kk - ci * KHW;
            int kh   = rem / KW;
            int kw   = rem - kh * KW;
            int cOff = ci * (Hin * Win);
            #pragma unroll
            for (int it = 0; it < 16; ++it) {
                int iy = (rowYX[it] >> 16) + kh;
                int ix = ((rowYX[it] << 16) >> 16) + kw;
                float v = 0.f;
                if (iy >= 0 && iy < Hin && ix >= 0 && ix < Win)
                    v = (float)in[rowBase[it] + cOff + iy * Win + ix];
                aReg[half][it] = (_Float16)v;
            }
        }
    };

    auto stageA = [&]() {
        #pragma unroll
        for (int half = 0; half < 2; ++half)
            #pragma unroll
            for (int it = 0; it < 16; ++it)
                As[it * 8 + wave][half * 32 + kc] = aReg[half][it];
    };

#if CDNA5_ASYNC
    auto asyncB = [&](int kBase, int b) {
        __builtin_amdgcn_global_load_async_to_lds_b128(
            (gv4i*)(wCol + kBase),
            (sv4i*)&Bt[b][bcol][bk], 0, 0);
        __builtin_amdgcn_global_load_async_to_lds_b128(
            (gv4i*)(wCol + kBase + 8),
            (sv4i*)&Bt[b][bcol][bk + 8], 0, 0);
    };
#else
    v8h bReg0, bReg1;
    auto fetchB = [&](int kBase) {
        bReg0 = *(const v8h*)(wCol + kBase);
        bReg1 = *(const v8h*)(wCol + kBase + 8);
    };
    auto stageB = [&](int b) {
        *(v8h*)&Bt[b][bcol][bk]     = bReg0;
        *(v8h*)&Bt[b][bcol][bk + 8] = bReg1;
    };
#endif

    v8f acc[4] = {};
    const int row = wave * 16 + l;
    int buf = 0;

#if CDNA5_ASYNC
    asyncB(0, 0);
#else
    fetchB(0);
#endif
    fetchA(0);

    for (int kBase = 0; kBase < Ktot; kBase += KC) {
        const bool more = (kBase + KC) < Ktot;
        stageA();
#if CDNA5_ASYNC
        if (more) {
            asyncB(kBase + KC, buf ^ 1);   // issue next tile into other buffer
            __builtin_amdgcn_s_wait_asynccnt(2);  // current tile's (older) ops done
        } else {
            __builtin_amdgcn_s_wait_asynccnt(0);
        }
#else
        stageB(buf);
        if (more) fetchB(kBase + KC);
#endif
        __syncthreads();
        if (more) fetchA(kBase + KC);      // global A loads overlap WMMA below

        #pragma unroll
        for (int half = 0; half < 2; ++half) {
            v8h a0 = *(const v8h*)&As[row][half * 32 + hi * 8];
            v8h a1 = *(const v8h*)&As[row][half * 32 + 16 + hi * 8];
            v16h a;
            #pragma unroll
            for (int e = 0; e < 8; ++e) { a[e] = a0[e]; a[8 + e] = a1[e]; }
            #pragma unroll
            for (int ns = 0; ns < 4; ++ns) {
                int col = ns * 16 + l;
                v8h b0 = *(const v8h*)&Bt[buf][col][half * 32 + hi * 8];
                v8h b1 = *(const v8h*)&Bt[buf][col][half * 32 + 16 + hi * 8];
                v16h b;
                #pragma unroll
                for (int e = 0; e < 8; ++e) { b[e] = b0[e]; b[8 + e] = b1[e]; }
                acc[ns] = __builtin_amdgcn_wmma_f32_16x16x32_f16(
                    false, a, false, b, (short)0, acc[ns], false, false);
            }
        }
        __syncthreads();
        buf ^= 1;
    }

    // ---- epilogue: C layout (VGPR r, lanes 0-15: M=r, N=lane; lanes 16-31: M=r+8) ----
    #pragma unroll
    for (int ns = 0; ns < 4; ++ns) {
        int oc = ocBase + ns * 16 + l;
        #pragma unroll
        for (int r = 0; r < 8; ++r) {
            int m = mBase + wave * 16 + hi * 8 + r;
            if (m < Mtotal) {
                int img = m / OHW;
                int p   = m - img * OHW;
                float v = acc[ns][r];
                if constexpr (GELU) v = gelu_f(v);
                out[((long long)img * OC + oc) * OHW + p] = (TOUT)v;
            }
        }
    }
}

// ---------------- VQ: argmin over 24 codes; emits zq(f16), z_probs, loss partials -----
__global__ __launch_bounds__(256) void vq_kernel(
    const float* __restrict__ z, const float* __restrict__ codes,
    const float* __restrict__ ema,
    _Float16* __restrict__ zq, float* __restrict__ zprobs,
    float* __restrict__ loss_acc, int B)
{
    const int C = 64, K = 24, HW = 63 * 63;
    __shared__ float cs[24 * 64];
    __shared__ float c2[24];
    __shared__ float ema_s[24];
    __shared__ float red[256];
    for (int i = threadIdx.x; i < K * C; i += blockDim.x) cs[i] = codes[i];
    if (threadIdx.x < K) {
        float s = 0.f;
        for (int c = 0; c < C; ++c) { float v = codes[threadIdx.x * C + c]; s += v * v; }
        c2[threadIdx.x]    = s;
        ema_s[threadIdx.x] = ema[threadIdx.x];
    }
    __syncthreads();

    int n = blockIdx.x * blockDim.x + threadIdx.x;
    const int N = B * HW;
    float localLoss = 0.f;
    if (n < N) {
        int img = n / HW;
        int p   = n - img * HW;
        const float* zp = z + (long long)img * C * HW + p;
        float zf[64];
        float z2 = 0.f;
        #pragma unroll
        for (int c = 0; c < C; ++c) { float v = zp[(long long)c * HW]; zf[c] = v; z2 += v * v; }
        int best = 0; float bestd = 3.4e38f;
        for (int k = 0; k < K; ++k) {
            float dot = 0.f;
            #pragma unroll
            for (int c = 0; c < C; ++c) dot += zf[c] * cs[k * C + c];
            float d = z2 - 2.f * dot + c2[k];
            if (d < bestd) { bestd = d; best = k; }
        }
        _Float16* zqp = zq + (long long)img * C * HW + p;
        #pragma unroll
        for (int c = 0; c < C; ++c) {
            float q  = cs[best * C + c];
            float df = q - zf[c];
            localLoss += df * df;
            zqp[(long long)c * HW] = (_Float16)q;
        }
        float esum = 0.f;
        for (int k = 0; k < K; ++k) esum += ema_s[k];
        zprobs[n] = ema_s[best] / esum;
    }
    red[threadIdx.x] = localLoss;
    __syncthreads();
    for (int s = 128; s > 0; s >>= 1) {
        if (threadIdx.x < s) red[threadIdx.x] += red[threadIdx.x + s];
        __syncthreads();
    }
    if (threadIdx.x == 0) atomicAdd(loss_acc, red[0]);
}

__global__ void init_kernel(float* loss_acc) {
    if (threadIdx.x == 0 && blockIdx.x == 0) *loss_acc = 0.f;
}
__global__ void finalize_kernel(const float* loss_acc, float* out_loss, float scale) {
    *out_loss = 0.25f * (*loss_acc) * scale;
}

extern "C" void kernel_launch(void* const* d_in, const int* in_sizes, int n_in,
                              void* d_out, int out_size, void* d_ws, size_t ws_size,
                              hipStream_t stream) {
    (void)in_sizes; (void)n_in; (void)out_size; (void)ws_size;
    const float* x      = (const float*)d_in[0];
    const float* enc_w1 = (const float*)d_in[1];
    const float* enc_w2 = (const float*)d_in[2];
    const float* enc_w3 = (const float*)d_in[3];
    const float* dec_w1 = (const float*)d_in[4];
    const float* dec_w2 = (const float*)d_in[5];
    const float* dec_w3 = (const float*)d_in[6];
    const float* codes  = (const float*)d_in[7];
    const float* ema    = (const float*)d_in[8];

    const int Bn = 32;
    float* out = (float*)d_out;
    const long long n_xhat = (long long)Bn * 256 * 64 * 64;  // 33554432
    const long long n_zp   = (long long)Bn * 63 * 63;        // 127008
    float* xhat   = out;
    float* zprobs = out + n_xhat;
    float* vqloss = out + n_xhat + n_zp;

    // workspace carve (256B-aligned bumps)
    char* wsp = (char*)d_ws;
    auto alloc = [&](size_t bytes) -> char* {
        char* p = wsp;
        wsp += (bytes + 255) & ~(size_t)255;
        return p;
    };
    _Float16* wB_e2 = (_Float16*)alloc(sizeof(_Float16) * 64  * 1600);
    _Float16* wB_e3 = (_Float16*)alloc(sizeof(_Float16) * 64  * 256);
    _Float16* wB_d1 = (_Float16*)alloc(sizeof(_Float16) * 512 * 256);
    _Float16* wB_d2 = (_Float16*)alloc(sizeof(_Float16) * 256 * 2048);
    _Float16* wB_d3 = (_Float16*)alloc(sizeof(_Float16) * 256 * 1024);
    _Float16* z1 = (_Float16*)alloc(sizeof(_Float16) * (size_t)Bn * 64 * 128 * 128);
    _Float16* z2 = (_Float16*)alloc(sizeof(_Float16) * (size_t)Bn * 64 * 64 * 64);
    float*    zf = (float*)   alloc(sizeof(float)    * (size_t)Bn * 64 * 63 * 63);
    _Float16* zq = (_Float16*)alloc(sizeof(_Float16) * (size_t)Bn * 64 * 63 * 63);
    _Float16* y1 = (_Float16*)alloc(sizeof(_Float16) * (size_t)Bn * 512 * 64 * 64);
    _Float16* y2 = (_Float16*)alloc(sizeof(_Float16) * (size_t)Bn * 256 * 63 * 63);
    float* loss_acc = (float*)alloc(256);

    auto cdiv = [](long long a, long long b) { return (int)((a + b - 1) / b); };

    init_kernel<<<1, 32, 0, stream>>>(loss_acc);

    wcast_kernel<<<cdiv(64 * 1600, 256), 256, 0, stream>>>(enc_w2, wB_e2, 64 * 1600);
    wcast_kernel<<<cdiv(64 * 256, 256),  256, 0, stream>>>(enc_w3, wB_e3, 64 * 256);
    wcast_kernel<<<cdiv(512 * 256, 256), 256, 0, stream>>>(dec_w1, wB_d1, 512 * 256);
    wcast_kernel<<<cdiv(256 * 2048, 256),256, 0, stream>>>(dec_w2, wB_d2, 256 * 2048);
    wcast_kernel<<<cdiv(256 * 1024, 256),256, 0, stream>>>(dec_w3, wB_d3, 256 * 1024);

    // enc1: direct conv + GELU
    conv1_kernel<<<cdiv((long long)Bn * 64 * 128 * 128, 256), 256, 0, stream>>>(x, enc_w1, z1, Bn);

    // enc2: [32,64,128,128] -> [32,64,64,64], 5x5 s2 p2, GELU. K=1600
    {
        dim3 g(cdiv((long long)Bn * 64 * 64, 128), 1);
        conv_wmma_kernel<5, 5, 2, 2, true, 64, 128, 128, 64, 64, 64, _Float16>
            <<<g, 256, 0, stream>>>(z1, wB_e2, z2, Bn);
    }
    // enc3: [32,64,64,64] -> [32,64,63,63] fp32, 2x2 s1 p0. K=256
    {
        dim3 g(cdiv((long long)Bn * 63 * 63, 128), 1);
        conv_wmma_kernel<2, 2, 1, 0, false, 64, 64, 64, 64, 63, 63, float>
            <<<g, 256, 0, stream>>>(z2, wB_e3, zf, Bn);
    }
    // VQ (z_hat == zq in forward by straight-through)
    vq_kernel<<<cdiv((long long)Bn * 63 * 63, 256), 256, 0, stream>>>(
        zf, codes, ema, zq, zprobs, loss_acc, Bn);
    // dec1: [32,64,63,63] -> [32,512,64,64], 2x2 p1, GELU. K=256
    {
        dim3 g(cdiv((long long)Bn * 64 * 64, 128), 512 / 64);
        conv_wmma_kernel<2, 2, 1, 1, true, 64, 63, 63, 512, 64, 64, _Float16>
            <<<g, 256, 0, stream>>>(zq, wB_d1, y1, Bn);
    }
    // dec2: [32,512,64,64] -> [32,256,63,63], 2x2 p0, GELU. K=2048
    {
        dim3 g(cdiv((long long)Bn * 63 * 63, 128), 256 / 64);
        conv_wmma_kernel<2, 2, 1, 0, true, 512, 64, 64, 256, 63, 63, _Float16>
            <<<g, 256, 0, stream>>>(y1, wB_d2, y2, Bn);
    }
    // dec3: [32,256,63,63] -> [32,256,64,64] fp32 into d_out, 2x2 p1. K=1024
    {
        dim3 g(cdiv((long long)Bn * 64 * 64, 128), 256 / 64);
        conv_wmma_kernel<2, 2, 1, 1, false, 256, 63, 63, 256, 64, 64, float>
            <<<g, 256, 0, stream>>>(y2, wB_d3, xhat, Bn);
    }
    finalize_kernel<<<1, 1, 0, stream>>>(loss_acc, vqloss,
                                         1.0f / ((float)Bn * 64 * 63 * 63));
}
